// OneTransAttention_52321291600425
// MI455X (gfx1250) — matrix operand order repaired and checked by reading, not compiled, not run on hardware
//
#include <hip/hip_runtime.h>
#include <hip/hip_bf16.h>
#include <math.h>

#define D_MODEL   1024
#define NUM_HEADS 16
#define HEAD_DIM  64
#define BATCH     2
#define SEQ       2048
#define SCALE_F   0.125f      // HEAD_DIM^-0.5
#define NEG_BIG   (-1e30f)

typedef __attribute__((ext_vector_type(16))) __bf16        v16bf;
typedef __attribute__((ext_vector_type(8)))  float         v8f;
typedef __attribute__((ext_vector_type(4)))  unsigned int  v4u;
typedef __attribute__((ext_vector_type(8)))  int           v8i;
typedef __attribute__((ext_vector_type(4)))  int           v4i;

// ---------------------------------------------------------------------------
// Tensor Data Mover: DMA a 2D bf16 tile (tile_w x tile_h) from global memory
// (row stride = row_stride_elems) into LDS at lds_off.  LDS padding is used so
// each row lands with an extra pad of 8 bf16 (4 DWORDs):
//   pad_icode: 3 -> pad every 16 DWORDs (32 bf16 row), 4 -> every 32 DWORDs
//   (64 bf16 row); pad_acode 3 -> 4 DWORDs (8 bf16).
// D# packing per CDNA5 ISA 8.3/8.4 (group0: count/lds/global/type,
// group1: data_size, pads, dims, stride).
// ---------------------------------------------------------------------------
__device__ __forceinline__ void tdm_load_2d_bf16(const __bf16* gsrc,
                                                 unsigned lds_off,
                                                 unsigned tile_w, unsigned tile_h,
                                                 unsigned row_stride_elems,
                                                 unsigned pad_icode,
                                                 unsigned pad_acode) {
  unsigned long long ga = (unsigned long long)(size_t)gsrc;
  v4u g0;
  g0[0] = 1u;                                       // count=1, no gather
  g0[1] = lds_off;                                  // LDS byte address
  g0[2] = (unsigned)(ga & 0xFFFFFFFFu);             // global_addr[31:0]
  g0[3] = (unsigned)((ga >> 32) & 0x01FFFFFFu)      // global_addr[56:32]
          | (2u << 30);                             // type=2 ("image")
  v8i g1;
  unsigned w0 = (1u << 16)                          // data_size = 2 bytes
              | (1u << 20)                          // pad_enable
              | (pad_icode << 22)                   // pad_interval
              | (pad_acode << 25);                  // pad_amount
  g1[0] = (int)w0;                                  // wg_mask=0 (not in cluster)
  g1[1] = (int)((tile_w & 0xFFFFu) << 16);          // tensor_dim0 = tile_w
  g1[2] = (int)((tile_h & 0xFFFFu) << 16);          // tensor_dim1 = tile_h
  g1[3] = (int)((tile_w & 0xFFFFu) << 16);          // tile_dim0
  g1[4] = (int)(tile_h & 0xFFFFu);                  // tile_dim1 (tile_dim2=0)
  g1[5] = (int)row_stride_elems;                    // tensor_dim0_stride[31:0]
  g1[6] = 0;                                        // stride hi / dim1_stride
  g1[7] = 0;
  v4i z = {};
#if defined(__clang_major__) && (__clang_major__ >= 23)
  __builtin_amdgcn_tensor_load_to_lds(g0, g1, z, z, (v8i){}, 0);
#else
  __builtin_amdgcn_tensor_load_to_lds(g0, g1, z, z, 0);
#endif
}

// ---------------------------------------------------------------------------
// fp32 -> bf16 conversion (grid-stride)
// ---------------------------------------------------------------------------
__global__ void f32_to_bf16_kernel(const float* __restrict__ in,
                                   __bf16* __restrict__ out, int n) {
  int i = blockIdx.x * blockDim.x + threadIdx.x;
  int stride = gridDim.x * blockDim.x;
  for (; i < n; i += stride) out[i] = (__bf16)in[i];
}

// ---------------------------------------------------------------------------
// Shared GEMM body: 64x64 tile / 128-thread block (4 waves), K-step 32.
// A tile DMA'd by TDM into As[64][40]; B tile staged TRANSPOSED into
// BsT[64][40] (uint4 global loads, b16 LDS scatter) so fragments are
// contiguous ds_load_b128 pairs.
// ---------------------------------------------------------------------------
#define GEMM_BODY(A, Bm, K, N, ACC)                                           \
  __shared__ __attribute__((aligned(16))) __bf16 As[64][40];                  \
  __shared__ __attribute__((aligned(16))) __bf16 BsT[64][40];                 \
  const int tid  = threadIdx.x;                                               \
  const int wave = tid >> 5, lane = tid & 31;                                 \
  const int lane15 = lane & 15;                                               \
  const bool hiHalf = lane >= 16;                                             \
  const int kbA = hiHalf ? 8 : 0;                                             \
  const int kbB = hiHalf ? 16 : 0;                                            \
  const int m0 = blockIdx.x * 64;                                             \
  const int n0 = blockIdx.y * 64;                                             \
  const unsigned lds_as = (unsigned)(size_t)(void*)&As[0][0];                 \
  v8f ACC[4];                                                                 \
  _Pragma("unroll") for (int t = 0; t < 4; ++t) ACC[t] = (v8f){};             \
  for (int kt = 0; kt < K; kt += 32) {                                        \
    if (wave == 0)                                                            \
      tdm_load_2d_bf16(A + (size_t)m0 * K + kt, lds_as, 32, 64, K, 3, 3);     \
    for (int i = tid; i < 256; i += 128) {                                    \
      int r = i >> 3, c = (i & 7) * 8;                                        \
      uint4 w = *(const uint4*)(Bm + (size_t)(kt + r) * N + n0 + c);          \
      const __bf16* e = (const __bf16*)&w;                                    \
      _Pragma("unroll") for (int j = 0; j < 8; ++j) BsT[c + j][r] = e[j];     \
    }                                                                         \
    if (kt + 32 < K)                                                          \
      __builtin_prefetch(Bm + (size_t)(kt + 32 + (tid >> 2)) * N + n0, 0, 0); \
    if (wave == 0) __builtin_amdgcn_s_wait_tensorcnt(0);                      \
    __syncthreads();                                                          \
    const int mrow = wave * 16 + lane15;                                      \
    v16bf af;                                                                 \
    _Pragma("unroll") for (int i = 0; i < 8; ++i) {                           \
      af[i]     = As[mrow][kbA + i];                                          \
      af[8 + i] = As[mrow][kbA + 16 + i];                                     \
    }                                                                         \
    _Pragma("unroll") for (int t = 0; t < 4; ++t) {                           \
      v16bf bfv;                                                              \
      const int col = t * 16 + lane15;                                        \
      _Pragma("unroll") for (int i = 0; i < 16; ++i)                          \
        bfv[i] = BsT[col][kbA + (i & 7) + ((i >> 3) << 4)];                   \
      ACC[t] = __builtin_amdgcn_wmma_f32_16x16x32_bf16(                       \
          false, af, false, bfv, (short)0, ACC[t], false, false);             \
    }                                                                         \
    __syncthreads();                                                          \
  }

// ---------------------------------------------------------------------------
// QKV projection GEMM; epilogue scatters into per-head Q/K/V [B,H,T,64] bf16.
// ---------------------------------------------------------------------------
__global__ __launch_bounds__(128)
void qkv_gemm_kernel(const __bf16* __restrict__ A, const __bf16* __restrict__ Bm,
                     const float* __restrict__ bias,
                     __bf16* __restrict__ Qb, __bf16* __restrict__ Kb,
                     __bf16* __restrict__ Vb) {
  GEMM_BODY(A, Bm, D_MODEL, (3 * D_MODEL), acc)
  const int which = n0 / D_MODEL;
  const int hcol  = (n0 % D_MODEL) / HEAD_DIM;
  __bf16* dst = (which == 0) ? Qb : (which == 1) ? Kb : Vb;
#pragma unroll
  for (int t = 0; t < 4; ++t) {
    const int n  = n0 + t * 16 + lane15;
    const int dd = t * 16 + lane15;
    const float bv = bias[n];
#pragma unroll
    for (int r = 0; r < 8; ++r) {
      const int m  = m0 + wave * 16 + r + (hiHalf ? 8 : 0);
      const int b  = m / SEQ, tq = m % SEQ;
      dst[(((size_t)b * NUM_HEADS + hcol) * SEQ + tq) * HEAD_DIM + dd] =
          (__bf16)(acc[t][r] + bv);
    }
  }
}

// ---------------------------------------------------------------------------
// Output projection GEMM -> f32 out
// ---------------------------------------------------------------------------
__global__ __launch_bounds__(128)
void out_gemm_kernel(const __bf16* __restrict__ A, const __bf16* __restrict__ Bm,
                     const float* __restrict__ bias, float* __restrict__ out) {
  GEMM_BODY(A, Bm, D_MODEL, D_MODEL, acc)
#pragma unroll
  for (int t = 0; t < 4; ++t) {
    const int n = n0 + t * 16 + lane15;
    const float bv = bias[n];
#pragma unroll
    for (int r = 0; r < 8; ++r) {
      const int m = m0 + wave * 16 + r + (hiHalf ? 8 : 0);
      out[(size_t)m * D_MODEL + n] = acc[t][r] + bv;
    }
  }
}

// ---------------------------------------------------------------------------
// Flash attention: one block per (b, h, 64 q rows).  Q/K tiles via TDM,
// V staged transposed, online softmax with mask, QK^T / PV on WMMA.
// ---------------------------------------------------------------------------
__global__ __launch_bounds__(128)
void attn_kernel(const __bf16* __restrict__ Qb, const __bf16* __restrict__ Kb,
                 const __bf16* __restrict__ Vb,
                 const unsigned char* __restrict__ mask,
                 __bf16* __restrict__ attnb) {
  __shared__ __attribute__((aligned(16))) __bf16 Qs[64][72];
  __shared__ __attribute__((aligned(16))) __bf16 Ks[32][72];
  __shared__ __attribute__((aligned(16))) __bf16 VsT[64][40];  // [dd][key]
  __shared__ __attribute__((aligned(16))) __bf16 Ps[4][16][40];

  const int tid  = threadIdx.x;
  const int wave = tid >> 5, lane = tid & 31;
  const int lane15 = lane & 15;
  const bool hiHalf = lane >= 16;
  const int kbA = hiHalf ? 8  : 0;
  const int kbB = hiHalf ? 16 : 0;
  const int bh = blockIdx.y;
  const int b  = bh >> 4;
  const int h  = bh & 15;
  const int q0 = blockIdx.x * 64;

  const __bf16* Qh = Qb + (size_t)bh * SEQ * HEAD_DIM;
  const __bf16* Kh = Kb + (size_t)bh * SEQ * HEAD_DIM;
  const __bf16* Vh = Vb + (size_t)bh * SEQ * HEAD_DIM;
  const unsigned char* mb = mask + (size_t)b * SEQ * SEQ;

  const unsigned lds_qs = (unsigned)(size_t)(void*)&Qs[0][0];
  const unsigned lds_ks = (unsigned)(size_t)(void*)&Ks[0][0];

  if (wave == 0) {
    tdm_load_2d_bf16(Qh + (size_t)q0 * HEAD_DIM, lds_qs, 64, 64, HEAD_DIM, 4, 3);
    __builtin_amdgcn_s_wait_tensorcnt(0);
  }
  __syncthreads();

  const int mrowq = wave * 16 + lane15;
  v16bf qf[2];
#pragma unroll
  for (int c2 = 0; c2 < 2; ++c2)
#pragma unroll
    for (int i = 0; i < 8; ++i) {
      qf[c2][i]     = Qs[mrowq][c2 * 32 + kbA + i];
      qf[c2][8 + i] = Qs[mrowq][c2 * 32 + kbA + 16 + i];
    }

  float mrun[8], lrun[8];
  v8f Oacc[4];
#pragma unroll
  for (int r = 0; r < 8; ++r) { mrun[r] = NEG_BIG; lrun[r] = 0.f; }
#pragma unroll
  for (int t = 0; t < 4; ++t) Oacc[t] = (v8f){};

  for (int kblk = 0; kblk < SEQ; kblk += 32) {
    if (wave == 0)
      tdm_load_2d_bf16(Kh + (size_t)kblk * HEAD_DIM, lds_ks, 64, 32, HEAD_DIM, 4, 3);
    // V tile transposed: coalesced uint4 global loads, b16 LDS scatter
    for (int i = tid; i < 256; i += 128) {
      int r = i >> 3, c = (i & 7) * 8;
      uint4 w = *(const uint4*)(Vh + (size_t)(kblk + r) * HEAD_DIM + c);
      const __bf16* e = (const __bf16*)&w;
#pragma unroll
      for (int j = 0; j < 8; ++j) VsT[c + j][r] = e[j];
    }
    if (wave == 0) __builtin_amdgcn_s_wait_tensorcnt(0);
    __syncthreads();

    // S = Q K^T : two 16-key subtiles, contraction over d=64 (2 chunks)
    v8f S[2];
#pragma unroll
    for (int t = 0; t < 2; ++t) {
      v8f s = (v8f){};
      const int keyrow = t * 16 + lane15;
#pragma unroll
      for (int c2 = 0; c2 < 2; ++c2) {
        v16bf kf;
#pragma unroll
        for (int i = 0; i < 16; ++i) kf[i] = Ks[keyrow][c2 * 32 + kbA + (i & 7) + ((i >> 3) << 4)];
        s = __builtin_amdgcn_wmma_f32_16x16x32_bf16(
            false, qf[c2], false, kf, (short)0, s, false, false);
      }
      S[t] = s;
    }

    // masked online softmax (row stats via width-16 xor shuffles)
    float p0v[8], p1v[8];
#pragma unroll
    for (int r = 0; r < 8; ++r) {
      const int qrow = q0 + wave * 16 + r + (hiHalf ? 8 : 0);
      const unsigned char mv0 = mb[(size_t)qrow * SEQ + kblk + lane15];
      const unsigned char mv1 = mb[(size_t)qrow * SEQ + kblk + 16 + lane15];
      const float s0 = mv0 ? S[0][r] * SCALE_F : NEG_BIG;
      const float s1 = mv1 ? S[1][r] * SCALE_F : NEG_BIG;
      float vmax = fmaxf(s0, s1);
#pragma unroll
      for (int off = 8; off >= 1; off >>= 1)
        vmax = fmaxf(vmax, __shfl_xor(vmax, off, 16));
      const float mnew = fmaxf(mrun[r], vmax);
      const float corr = __expf(mrun[r] - mnew);
      const float p0 = mv0 ? __expf(s0 - mnew) : 0.f;
      const float p1 = mv1 ? __expf(s1 - mnew) : 0.f;
      float psum = p0 + p1;
#pragma unroll
      for (int off = 8; off >= 1; off >>= 1) psum += __shfl_xor(psum, off, 16);
      lrun[r] = lrun[r] * corr + psum;
      mrun[r] = mnew;
#pragma unroll
      for (int t = 0; t < 4; ++t) Oacc[t][r] *= corr;
      p0v[r] = p0; p1v[r] = p1;
    }

    // P (C-layout) -> WMMA-A layout via per-wave LDS scratch
    const int prow = hiHalf ? 8 : 0;
#pragma unroll
    for (int r = 0; r < 8; ++r) {
      Ps[wave][r + prow][lane15]      = (__bf16)p0v[r];
      Ps[wave][r + prow][16 + lane15] = (__bf16)p1v[r];
    }
    asm volatile("s_wait_dscnt 0" ::: "memory");  // intra-wave LDS RAW fence
    v16bf pf;
#pragma unroll
    for (int i = 0; i < 8; ++i) {
      pf[i]     = Ps[wave][lane15][kbA + i];
      pf[8 + i] = Ps[wave][lane15][kbA + 16 + i];
    }

    // O += P V  (contraction over 32 keys; V fragments contiguous from VsT)
#pragma unroll
    for (int td = 0; td < 4; ++td) {
      v16bf vf;
      const int col = td * 16 + lane15;
#pragma unroll
      for (int i = 0; i < 16; ++i) vf[i] = VsT[col][kbA + (i & 7) + ((i >> 3) << 4)];
      Oacc[td] = __builtin_amdgcn_wmma_f32_16x16x32_bf16(
          false, pf, false, vf, (short)0, Oacc[td], false, false);
    }
    __syncthreads();
  }

  // normalize and write [B,T,D] bf16 (rows with no valid key -> 0)
#pragma unroll
  for (int td = 0; td < 4; ++td) {
#pragma unroll
    for (int r = 0; r < 8; ++r) {
      const int qrow = q0 + wave * 16 + r + (hiHalf ? 8 : 0);
      const float denom = lrun[r];
      const float val = (denom > 0.f) ? (Oacc[td][r] / denom) : 0.f;
      attnb[((size_t)b * SEQ + qrow) * D_MODEL + h * HEAD_DIM + td * 16 + lane15] =
          (__bf16)val;
    }
  }
}

// ---------------------------------------------------------------------------
extern "C" void kernel_launch(void* const* d_in, const int* in_sizes, int n_in,
                              void* d_out, int out_size, void* d_ws, size_t ws_size,
                              hipStream_t stream) {
  (void)in_sizes; (void)n_in; (void)out_size; (void)ws_size;
  const float*         x     = (const float*)d_in[0];
  const unsigned char* mask  = (const unsigned char*)d_in[1];  // jax bool = 1B
  const float*         W_qkv = (const float*)d_in[2];
  const float*         b_qkv = (const float*)d_in[3];
  const float*         W_out = (const float*)d_in[4];
  const float*         b_out = (const float*)d_in[5];
  float* out = (float*)d_out;

  char* ws = (char*)d_ws;
  size_t off = 0;
  auto alloc_bf16 = [&](size_t elems) {
    __bf16* p = (__bf16*)(ws + off);
    off += elems * sizeof(__bf16);
    return p;
  };
  const size_t MT = (size_t)BATCH * SEQ;                      // 4096
  __bf16* xb    = alloc_bf16(MT * D_MODEL);                   // 8 MB
  __bf16* wqkvb = alloc_bf16((size_t)D_MODEL * 3 * D_MODEL);  // 6 MB
  __bf16* woutb = alloc_bf16((size_t)D_MODEL * D_MODEL);      // 2 MB
  __bf16* Qb    = alloc_bf16(MT * D_MODEL);                   // 8 MB
  __bf16* Kb    = alloc_bf16(MT * D_MODEL);                   // 8 MB
  __bf16* Vb    = alloc_bf16(MT * D_MODEL);                   // 8 MB
  __bf16* attnb = alloc_bf16(MT * D_MODEL);                   // 8 MB

  f32_to_bf16_kernel<<<2048, 256, 0, stream>>>(x, xb, (int)(MT * D_MODEL));
  f32_to_bf16_kernel<<<2048, 256, 0, stream>>>(W_qkv, wqkvb, D_MODEL * 3 * D_MODEL);
  f32_to_bf16_kernel<<<1024, 256, 0, stream>>>(W_out, woutb, D_MODEL * D_MODEL);

  dim3 g1((unsigned)(MT / 64), (3 * D_MODEL) / 64);           // 64 x 48
  qkv_gemm_kernel<<<g1, 128, 0, stream>>>(xb, wqkvb, b_qkv, Qb, Kb, Vb);

  dim3 g2(SEQ / 64, BATCH * NUM_HEADS);                       // 32 x 32
  attn_kernel<<<g2, 128, 0, stream>>>(Qb, Kb, Vb, mask, attnb);

  dim3 g3((unsigned)(MT / 64), D_MODEL / 64);                 // 64 x 16
  out_gemm_kernel<<<g3, 128, 0, stream>>>(attnb, woutb, b_out, out);
}
